// CausalSelfAttention_72885595013565
// MI455X (gfx1250) — matrix operand order repaired and checked by reading, not compiled, not run on hardware
//
#include <hip/hip_runtime.h>

#define SEQ_T 2048
#define MODEL_DIM 2048
#define ATTN_SCALE_F 0.12f

typedef __attribute__((ext_vector_type(16))) __bf16    v16bf;
typedef __attribute__((ext_vector_type(8)))  float     v8f;
typedef __attribute__((ext_vector_type(4)))  int       v4i_t;
typedef __attribute__((ext_vector_type(4)))  unsigned  v4u_t;
typedef __attribute__((ext_vector_type(8)))  int       v8i_t;

// ---- CDNA5 async global->LDS copy (ASYNCcnt path), with safe fallback ----
#if defined(__has_builtin)
#  if __has_builtin(__builtin_amdgcn_global_load_async_to_lds_b128)
#    define HAVE_ASYNC_LDS 1
#  endif
#  if __has_builtin(__builtin_amdgcn_tensor_load_to_lds)
#    define HAVE_TDM 1
#  endif
#endif
#ifndef HAVE_ASYNC_LDS
#  define HAVE_ASYNC_LDS 0
#endif
#ifndef HAVE_TDM
#  define HAVE_TDM 0
#endif

typedef __attribute__((address_space(1))) v4i_t* g128_t;   // global int4*
typedef __attribute__((address_space(3))) v4i_t* l128_t;   // LDS int4*

#if HAVE_ASYNC_LDS
#  define ASYNC_COPY16(g, l)                                                  \
    __builtin_amdgcn_global_load_async_to_lds_b128((g128_t)(g), (l128_t)(l), 0, 0)
#  define ASYNC_WAIT() asm volatile("s_wait_asynccnt 0x0" ::: "memory")
#else
#  define ASYNC_COPY16(g, l) (*(uint4*)(l) = *(const uint4*)(g))
#  define ASYNC_WAIT() do { } while (0)
#endif

__device__ __forceinline__ unsigned lds_byte_off(const void* p) {
  return (unsigned)(unsigned long long)(l128_t)p;   // AS3 ptr value = LDS byte offset
}

#if HAVE_TDM
// TENSOR_LOAD_TO_LDS of a 2D bf16 tile: tile_k elems wide, tile_rows rows, with
// hardware LDS padding of 16B after every 64B (-> pitch 40 halves).
__device__ __forceinline__ void tdm_load_tile(const void* gaddr, unsigned lds_off,
                                              unsigned tile_k, unsigned tile_rows,
                                              unsigned row_stride_elems) {
  unsigned long long ga = (unsigned long long)gaddr;
  v4u_t g0;
  g0.x = 1u;                                   // count=1, user descriptor
  g0.y = lds_off;                              // D#.lds_addr (bytes)
  g0.z = (unsigned)(ga & 0xFFFFFFFFu);         // global_addr[31:0]
  g0.w = (unsigned)((ga >> 32) & 0x1FFFFFFu) | (2u << 30);  // addr[56:32] | type=2
  v8i_t g1;
  g1[0] = (int)((1u << 16)      // data_size = 1 -> 2 bytes
              | (1u << 20)      // pad_enable
              | (3u << 22)      // pad_interval: 16 DWORDs (64B) before padding
              | (3u << 25));    // pad_amount:   4 DWORDs (16B)
  g1[1] = (int)((MODEL_DIM & 0xFFFFu) << 16);                    // tensor_dim0[15:0]
  g1[2] = (int)(((MODEL_DIM >> 16) & 0xFFFFu) |
                ((SEQ_T & 0xFFFFu) << 16));                      // dim0 hi | tensor_dim1 lo
  g1[3] = (int)(((SEQ_T >> 16) & 0xFFFFu) | ((tile_k & 0xFFFFu) << 16)); // dim1 hi | tile_dim0
  g1[4] = (int)(tile_rows & 0xFFFFu);                            // tile_dim1 (tile_dim2=0)
  g1[5] = (int)row_stride_elems;                                 // tensor_dim0_stride[31:0]
  g1[6] = 0;
  g1[7] = 0;
  v4i_t gz4 = {0, 0, 0, 0};
  v8i_t gz8 = {0, 0, 0, 0, 0, 0, 0, 0};
  // 6-arg toolchain form: (g0, g1, g2, g3, <extended group>, cpol)
  __builtin_amdgcn_tensor_load_to_lds(g0, g1, gz4, gz4, gz8, 0);
}
#endif

__device__ __forceinline__ unsigned short f2bf(float f) {
  unsigned int u = __float_as_uint(f);
  unsigned int r = u + 0x7FFFu + ((u >> 16) & 1u);   // round to nearest even
  return (unsigned short)(r >> 16);
}
__device__ __forceinline__ float bf2f(unsigned short h) {
  return __uint_as_float(((unsigned int)h) << 16);
}

// ---- WMMA fragment loaders (CDNA5 16-bit layouts, ISA 7.12.2) ----
__device__ __forceinline__ v16bf ld_afrag(const unsigned short* lds, int pitch,
                                          int mbase, int kk, int lane) {
  const unsigned short* p = lds + (mbase + (lane & 15)) * pitch + kk + ((lane >> 4) << 3);
  union { uint4 q[2]; v16bf v; } u;
  u.q[0] = *(const uint4*)p;
  u.q[1] = *(const uint4*)(p + 16);
  return u.v;
}
__device__ __forceinline__ v16bf ld_bfrag(const unsigned short* lds, int pitch,
                                          int nbase, int kk, int lane) {
  const unsigned short* p = lds + (nbase + (lane & 15)) * pitch + kk + ((lane >> 4) << 3);
  union { uint4 q[2]; v16bf v; } u;
  u.q[0] = *(const uint4*)p;
  u.q[1] = *(const uint4*)(p + 16);
  return u.v;
}
__device__ __forceinline__ v8f wmma_bf(v16bf a, v16bf b, v8f c) {
  return __builtin_amdgcn_wmma_f32_16x16x32_bf16(false, a, false, b, (short)0, c,
                                                 false, false);
}

// ---------------- fp32 -> bf16 conversion ----------------
__global__ __launch_bounds__(256) void cvt_bf16(const float* __restrict__ src,
                                                unsigned short* __restrict__ dst, int n4) {
  for (int i = blockIdx.x * blockDim.x + threadIdx.x; i < n4; i += gridDim.x * blockDim.x) {
    float4 f = ((const float4*)src)[i];
    uint2 o;
    o.x = (unsigned)f2bf(f.x) | ((unsigned)f2bf(f.y) << 16);
    o.y = (unsigned)f2bf(f.z) | ((unsigned)f2bf(f.w) << 16);
    ((uint2*)dst)[i] = o;
  }
}

// ---------------- TN GEMM: C[m,n] = sum_k A[m,k]*B[n,k], 2048x2048x2048 ----------------
// Double-buffered LDS; TDM (or async-LDS) staging; K-slab = 32, pitch 40 (conflict-free).
template <bool OUTF32>
__global__ __launch_bounds__(256) void gemm_tn_bf16(const unsigned short* __restrict__ A,
                                                    const unsigned short* __restrict__ Bmat,
                                                    void* __restrict__ Cv,
                                                    long strideB, long strideC) {
  __shared__ unsigned short As[2][128 * 40];
  __shared__ unsigned short Bs[2][128 * 40];
  const unsigned short* B = Bmat + (long)blockIdx.z * strideB;
  const int tid  = threadIdx.x;
  const int lane = tid & 31;
  const int wave = tid >> 5;
  const int wm = (wave >> 1) * 32;
  const int wn = (wave & 1) * 64;
  const int m0 = blockIdx.x * 128;
  const int n0 = blockIdx.y * 128;

  auto stage = [&](int buf, int k0) {
#if HAVE_TDM
    if (wave == 0) {   // TDM ignores EXEC; one issue per wave => one DMA per tile
      tdm_load_tile(&A[(long)m0 * MODEL_DIM + k0], lds_byte_off(&As[buf][0]),
                    32, 128, MODEL_DIM);
      tdm_load_tile(&B[(long)n0 * MODEL_DIM + k0], lds_byte_off(&Bs[buf][0]),
                    32, 128, MODEL_DIM);
    }
#else
    unsigned short* sA = &As[buf][0];
    unsigned short* sB = &Bs[buf][0];
    for (int i = tid; i < 512; i += 256) {
      int row = i >> 2, ck = (i & 3) << 3;
      ASYNC_COPY16(&A[(long)(m0 + row) * MODEL_DIM + k0 + ck], &sA[row * 40 + ck]);
      ASYNC_COPY16(&B[(long)(n0 + row) * MODEL_DIM + k0 + ck], &sB[row * 40 + ck]);
    }
#endif
  };

  v8f z = {0.f, 0.f, 0.f, 0.f, 0.f, 0.f, 0.f, 0.f};
  v8f acc[2][4];
  for (int i = 0; i < 2; ++i)
    for (int j = 0; j < 4; ++j) acc[i][j] = z;

  stage(0, 0);
  for (int k0 = 0; k0 < MODEL_DIM; k0 += 32) {
    const int cur = (k0 >> 5) & 1;
#if HAVE_TDM
    if (wave == 0) __builtin_amdgcn_s_wait_tensorcnt(0);  // issuer waits its DMAs
#else
    ASYNC_WAIT();
#endif
    __syncthreads();         // all threads see slab `cur` resident
    if (k0 + 32 < MODEL_DIM) {
      stage(cur ^ 1, k0 + 32);
      if (k0 + 64 < MODEL_DIM) {  // pull the slab after next toward L2
        __builtin_prefetch(&A[(long)(m0 + (tid >> 1)) * MODEL_DIM + k0 + 64], 0, 1);
        __builtin_prefetch(&B[(long)(n0 + (tid >> 1)) * MODEL_DIM + k0 + 64], 0, 1);
      }
    }
    const unsigned short* sA = &As[cur][0];
    const unsigned short* sB = &Bs[cur][0];
    v16bf af[2], bfv[4];
    for (int mi = 0; mi < 2; ++mi) af[mi] = ld_afrag(sA, 40, wm + mi * 16, 0, lane);
    for (int ni = 0; ni < 4; ++ni) bfv[ni] = ld_bfrag(sB, 40, wn + ni * 16, 0, lane);
    for (int mi = 0; mi < 2; ++mi)
      for (int ni = 0; ni < 4; ++ni)
        acc[mi][ni] = wmma_bf(af[mi], bfv[ni], acc[mi][ni]);
  }

  const int half = lane >> 4, cl = lane & 15;
  for (int mi = 0; mi < 2; ++mi)
    for (int ni = 0; ni < 4; ++ni)
      for (int r = 0; r < 8; ++r) {
        long m = m0 + wm + mi * 16 + r + 8 * half;
        long n = n0 + wn + ni * 16 + cl;
        float v = acc[mi][ni][r];
        if (OUTF32)
          ((float*)Cv + (long)blockIdx.z * strideC)[m * MODEL_DIM + n] = v;
        else
          ((unsigned short*)Cv + (long)blockIdx.z * strideC)[m * MODEL_DIM + n] = f2bf(v);
      }
}

// ---------------- RMS norm + RoPE + lambda blend (one wave per (t,head)) ----------------
__global__ __launch_bounds__(256) void norm_rope_lambda(unsigned short* __restrict__ qkvb,
                                                        const float* __restrict__ ve,
                                                        const float* __restrict__ lam) {
  const int lane = threadIdx.x & 31;
  const int wave = threadIdx.x >> 5;
  const int u = blockIdx.x * 8 + wave;
  const int t = u >> 4;
  const int h = u & 15;
  const float l0 = lam[0], l1 = lam[1];

  float fr  = exp2f(-10.0f * (float)lane / 31.0f);   // (1/1024)^(lane/31)
  float th  = (float)t * fr;
  float cth = __cosf(th), sth = __sinf(th);

  for (int e = 0; e < 3; ++e) {
    unsigned short* base = qkvb + ((long)e * SEQ_T + t) * MODEL_DIM + h * 128;
    float x[4];
    float ss = 0.f;
    for (int j = 0; j < 4; ++j) {
      x[j] = bf2f(base[lane + 32 * j]);
      ss += x[j] * x[j];
    }
    for (int m = 16; m >= 1; m >>= 1) ss += __shfl_xor(ss, m, 32);
    float inv = rsqrtf(ss * (1.0f / 128.0f) + 1e-6f);
    for (int j = 0; j < 4; ++j) x[j] *= inv;
    if (e < 2) {
      float x0 = x[0], x2 = x[2];
      x[0] =  x0 * cth + x2 * sth;
      x[2] = -x0 * sth + x2 * cth;
    } else {
      for (int j = 0; j < 4; ++j)
        x[j] = l0 * x[j] + l1 * ve[(long)t * MODEL_DIM + h * 128 + lane + 32 * j];
    }
    for (int j = 0; j < 4; ++j) base[lane + 32 * j] = f2bf(x[j]);
  }
}

// ---------------- flash attention: 64 q-rows per block, 4 waves x 16 rows ----------------
__global__ __launch_bounds__(128) void attn_fwd(const unsigned short* __restrict__ qkvb,
                                                unsigned short* __restrict__ yb) {
  __shared__ unsigned short Qs[64 * 136];
  __shared__ unsigned short Ks[64 * 136];
  __shared__ unsigned short Vt[128 * 72];   // V transposed: [c][s]
  __shared__ unsigned short Ps[4][16 * 72]; // per-wave P staging

  const int tid  = threadIdx.x;
  const int lane = tid & 31;
  const int wave = tid >> 5;
  const int half = lane >> 4;
  const int cl   = lane & 15;
  const int q0 = blockIdx.x * 64;
  const int h  = blockIdx.y;
  const int mw = wave * 16;

  const unsigned short* qp = qkvb;
  const unsigned short* kp = qkvb + (long)SEQ_T * MODEL_DIM;
  const unsigned short* vp = qkvb + 2L * SEQ_T * MODEL_DIM;

  for (int i = tid; i < 1024; i += 128) {
    int row = i >> 4, ck = (i & 15) << 3;
    ASYNC_COPY16(&qp[(long)(q0 + row) * MODEL_DIM + h * 128 + ck], &Qs[row * 136 + ck]);
  }

  v8f z = {0.f, 0.f, 0.f, 0.f, 0.f, 0.f, 0.f, 0.f};
  v8f yacc[8];
  for (int jc = 0; jc < 8; ++jc) yacc[jc] = z;
  float m_i[8], l_i[8];
  for (int r = 0; r < 8; ++r) { m_i[r] = -1e30f; l_i[r] = 0.f; }

  ASYNC_WAIT();
  __syncthreads();

  for (int s0 = 0; s0 <= q0; s0 += 64) {
    for (int i = tid; i < 1024; i += 128) {
      int row = i >> 4, ck = (i & 15) << 3;
      ASYNC_COPY16(&kp[(long)(s0 + row) * MODEL_DIM + h * 128 + ck], &Ks[row * 136 + ck]);
      union { uint4 q; unsigned short s[8]; } vv;   // V needs a transpose -> VGPR + scatter
      vv.q = *(const uint4*)&vp[(long)(s0 + row) * MODEL_DIM + h * 128 + ck];
      for (int j = 0; j < 8; ++j) Vt[(ck + j) * 72 + row] = vv.s[j];
    }
    ASYNC_WAIT();
    __syncthreads();

    v8f sf[4];
    for (int ni = 0; ni < 4; ++ni) {
      v8f s = z;
      for (int kk = 0; kk < 128; kk += 32)
        s = wmma_bf(ld_afrag(Qs, 136, mw, kk, lane),
                    ld_bfrag(Ks, 136, ni * 16, kk, lane), s);
      sf[ni] = s;
    }

    const bool diag = (s0 == q0);
    for (int ni = 0; ni < 4; ++ni)
      for (int r = 0; r < 8; ++r) {
        float v = sf[ni][r] * ATTN_SCALE_F;
        if (diag && (s0 + ni * 16 + cl) > (q0 + mw + r + 8 * half)) v = -3.0e38f;
        sf[ni][r] = v;
      }

    unsigned short* pw = &Ps[wave][0];
    for (int r = 0; r < 8; ++r) {
      float mx = sf[0][r];
      for (int ni = 1; ni < 4; ++ni) mx = fmaxf(mx, sf[ni][r]);
      for (int m = 8; m >= 1; m >>= 1) mx = fmaxf(mx, __shfl_xor(mx, m, 32));
      float mnew  = fmaxf(m_i[r], mx);
      float alpha = __expf(m_i[r] - mnew);
      float rs = 0.f;
      for (int ni = 0; ni < 4; ++ni) {
        float p = __expf(sf[ni][r] - mnew);
        sf[ni][r] = p;
        rs += p;
      }
      for (int m = 8; m >= 1; m >>= 1) rs += __shfl_xor(rs, m, 32);
      l_i[r] = l_i[r] * alpha + rs;
      m_i[r] = mnew;
      for (int jc = 0; jc < 8; ++jc) yacc[jc][r] *= alpha;
      int prow = r + 8 * half;
      for (int ni = 0; ni < 4; ++ni)
        pw[prow * 72 + ni * 16 + cl] = f2bf(sf[ni][r]);
    }
    asm volatile("s_wait_dscnt 0x0" ::: "memory");  // per-wave LDS RAW before A-frag reads

    for (int kk = 0; kk < 64; kk += 32) {
      v16bf pa = ld_afrag(pw, 72, 0, kk, lane);
      for (int jc = 0; jc < 8; ++jc)
        yacc[jc] = wmma_bf(pa, ld_bfrag(Vt, 72, jc * 16, kk, lane), yacc[jc]);
    }
    __syncthreads();
  }

  for (int jc = 0; jc < 8; ++jc)
    for (int r = 0; r < 8; ++r) {
      long t = q0 + mw + r + 8 * half;
      float yv = yacc[jc][r] / l_i[r];
      yb[t * MODEL_DIM + h * 128 + jc * 16 + cl] = f2bf(yv);
    }
}

extern "C" void kernel_launch(void* const* d_in, const int* in_sizes, int n_in,
                              void* d_out, int out_size, void* d_ws, size_t ws_size,
                              hipStream_t stream) {
  const float* x   = (const float*)d_in[0];
  const float* w   = (const float*)d_in[1];
  const float* ve  = (const float*)d_in[2];
  const float* lam = (const float*)d_in[3];

  const long NE = (long)MODEL_DIM * SEQ_T;
  unsigned short* xb   = (unsigned short*)d_ws;
  unsigned short* wb   = xb + NE;
  unsigned short* qkvb = wb + 4 * NE;
  unsigned short* yb   = qkvb + 3 * NE;

  cvt_bf16<<<1024, 256, 0, stream>>>(x, xb, (int)(NE / 4));
  cvt_bf16<<<2048, 256, 0, stream>>>(w, wb, (int)NE);

  gemm_tn_bf16<false><<<dim3(16, 16, 3), 256, 0, stream>>>(xb, wb, (void*)qkvb, NE, NE);
  norm_rope_lambda<<<4096, 256, 0, stream>>>(qkvb, ve, lam);
  attn_fwd<<<dim3(32, 16), 128, 0, stream>>>(qkvb, yb);
  gemm_tn_bf16<true><<<dim3(16, 16, 1), 256, 0, stream>>>(yb, wb + 3 * NE, d_out, 0, 0);
}
